// CLModel_40_1_01_73083163509635
// MI455X (gfx1250) — matrix operand order repaired and checked by reading, not compiled, not run on hardware
//
#include <hip/hip_runtime.h>
#include <hip/hip_bf16.h>

// ---------------- types ----------------
typedef __attribute__((ext_vector_type(16))) __bf16 v16bf;
typedef __attribute__((ext_vector_type(8)))  float  v8f;
typedef __attribute__((ext_vector_type(4)))  float  v4f;

#define B_   64
#define T_   2048
#define D_   256
#define U_   256
#define NG_  768                     // 3*U
#define NT_  48                      // NG_/16 n-tiles
#define KC_  8                       // D_/32 k-chunks
#define KLDS 5                       // k-chunks of R staged in LDS
#define FRAGE 16                     // bf16 elems per lane per fragment
#define WFRAG (NT_*KC_*32*FRAGE)     // 196608 elems per weight matrix

__device__ __forceinline__ v8f splat8(float v) {
    v8f r;
#pragma unroll
    for (int i = 0; i < 8; ++i) r[i] = v;
    return r;
}

__device__ __forceinline__ v8f sigmoid8(v8f x) {
    v8f r;
#pragma unroll
    for (int i = 0; i < 8; ++i) r[i] = 1.0f / (1.0f + __expf(-x[i]));
    return r;
}

__device__ __forceinline__ v8f tanh8(v8f x) {
    v8f r;
#pragma unroll
    for (int i = 0; i < 8; ++i) r[i] = tanhf(x[i]);
    return r;
}

// Build a bf16 A-fragment (16x32 tile slice for this lane) from an f32 row.
__device__ __forceinline__ v16bf load_ax(const float* __restrict__ p, int kc, int khalf) {
    v4f a = *(const v4f*)(p + kc * 32 + khalf * 8);
    v4f b = *(const v4f*)(p + kc * 32 + khalf * 8 + 4);
    v4f c = *(const v4f*)(p + kc * 32 + 16 + khalf * 8);
    v4f d = *(const v4f*)(p + kc * 32 + 16 + khalf * 8 + 4);
    v16bf r;
#pragma unroll
    for (int e = 0; e < 4; ++e) {
        r[e]      = (__bf16)a[e];
        r[4 + e]  = (__bf16)b[e];
        r[8 + e]  = (__bf16)c[e];
        r[12 + e] = (__bf16)d[e];
    }
    return r;
}

#define WMMA_BF16(Afrag, Bfrag, Cacc) \
    __builtin_amdgcn_wmma_f32_16x16x32_bf16(false, (Afrag), false, (Bfrag), (short)0, (Cacc), false, false)

// ---------------------------------------------------------------------------
// Kernel 1: repack W [256,768] and R [256,768] (f32) into bf16 WMMA
// B-fragment layout: fragment (nt,kc): lane<16 holds K=kc*32+0..7,16..23 (2/VGPR),
// N=nt*16+lane; lane>=16 holds K=kc*32+8..15,24..31, N=nt*16+(lane-16).
// ---------------------------------------------------------------------------
__global__ __launch_bounds__(256) void prep_weights(const float* __restrict__ W,
                                                    const float* __restrict__ R,
                                                    __bf16* __restrict__ wsW,
                                                    __bf16* __restrict__ wsR) {
    int idx = blockIdx.x * 256 + threadIdx.x;
    if (idx >= WFRAG) return;
    int e    = idx & 15;
    int lane = (idx >> 4) & 31;
    int kc   = (idx >> 9) & 7;
    int nt   = idx >> 12;
    int k = kc * 32 + (e >> 3) * 16 + (lane >> 4) * 8 + (e & 7);
    int n = nt * 16 + (lane & 15);
    wsW[idx] = (__bf16)W[k * NG_ + n];
    wsR[idx] = (__bf16)R[k * NG_ + n];
}

// ---------------------------------------------------------------------------
// Kernel 2: xw = x @ W + bias, massively parallel over all B*T rows.
// Block = 32 rows (2 M-tiles) x all 768 cols; wave w owns n-tiles w*6..w*6+5.
// ---------------------------------------------------------------------------
__global__ __launch_bounds__(256) void xw_gemm(const float* __restrict__ x,
                                               const float* __restrict__ bias,
                                               const __bf16* __restrict__ wsW,
                                               float* __restrict__ xw) {
    const int wave  = threadIdx.x >> 5;
    const int lane  = threadIdx.x & 31;
    const int khalf = lane >> 4;
    const int ncol  = lane & 15;
    const size_t r0 = (size_t)blockIdx.x * 32;

    v8f c[2][6];
#pragma unroll
    for (int j = 0; j < 6; ++j) {
        float bv = bias[(wave * 6 + j) * 16 + ncol];
        c[0][j] = splat8(bv);
        c[1][j] = splat8(bv);
    }

    const float* xr0 = x + (r0 + ncol) * (size_t)D_;        // M-tile 0 row data
    const float* xr1 = x + (r0 + 16 + ncol) * (size_t)D_;   // M-tile 1 row data

#pragma unroll 1
    for (int kc = 0; kc < KC_; ++kc) {
        v16bf ax0 = load_ax(xr0, kc, khalf);
        v16bf ax1 = load_ax(xr1, kc, khalf);
#pragma unroll
        for (int j = 0; j < 6; ++j) {
            int nt = wave * 6 + j;
            v16bf bwf = *(const v16bf*)(wsW + ((size_t)(nt * KC_ + kc) * 32 + lane) * FRAGE);
            c[0][j] = WMMA_BF16(ax0, bwf, c[0][j]);
            c[1][j] = WMMA_BF16(ax1, bwf, c[1][j]);
        }
    }

#pragma unroll
    for (int m = 0; m < 2; ++m)
#pragma unroll
        for (int j = 0; j < 6; ++j)
#pragma unroll
            for (int i = 0; i < 8; ++i) {
                size_t row = r0 + m * 16 + i + 8 * khalf;
                xw[row * NG_ + (wave * 6 + j) * 16 + ncol] = c[m][j][i];
            }
}

// ---------------------------------------------------------------------------
// Kernel 3: GRU scan. 4 persistent workgroups (16 batch rows), 8 waves; wave
// w owns features [32w,32w+32) => matching z/r/hhat gate tiles, so the state
// update is wave-local. R k-chunks 0..4 staged in LDS (240 KB); k-chunks 5..7
// streamed from L2 each step. FUSED=true additionally computes x_t @ W inline
// (fallback when workspace is too small for xw).
// An opaque zero (tz) is folded into every weight-fragment address so LICM
// cannot hoist the (t-invariant) fragment loads out of the time loop.
// ---------------------------------------------------------------------------
template <bool FUSED>
__global__ __launch_bounds__(256) void gru_scan(const float* __restrict__ x,
                                                const float* __restrict__ bias,
                                                const __bf16* __restrict__ wsW,
                                                const __bf16* __restrict__ wsR,
                                                const float* __restrict__ xw,
                                                float* __restrict__ out) {
    extern __shared__ __align__(32) char smem[];
    __bf16* hbuf = (__bf16*)smem;                    // 8 KB: h in A-fragment layout
    __bf16* ldsR = (__bf16*)(smem + 8192);           // 240 KB: R k-chunks 0..4

    const int wave  = threadIdx.x >> 5;
    const int lane  = threadIdx.x & 31;
    const int khalf = lane >> 4;
    const int ncol  = lane & 15;
    const int b0    = blockIdx.x * 16;
    const int f0    = wave * 32;

    // stage R (kc < KLDS) into LDS, 32B per copy
    for (int fi = threadIdx.x; fi < NT_ * KLDS * 32; fi += 256) {
        int nt  = fi / (KLDS * 32);
        int rem = fi - nt * (KLDS * 32);
        int kc  = rem >> 5;
        int ln  = rem & 31;
        *(v16bf*)&ldsR[(size_t)fi * FRAGE] =
            *(const v16bf*)&wsR[((size_t)(nt * KC_ + kc) * 32 + ln) * FRAGE];
    }
    // zero h state
    for (int i = threadIdx.x; i < KC_ * 32 * FRAGE; i += 256) hbuf[i] = (__bf16)0.0f;
    v8f hold0 = splat8(0.0f);
    v8f hold1 = splat8(0.0f);
    __syncthreads();

    float biasv[6];
#pragma unroll
    for (int j = 0; j < 2; ++j) {
        biasv[j]     = bias[f0 + j * 16 + ncol];
        biasv[2 + j] = bias[256 + f0 + j * 16 + ncol];
        biasv[4 + j] = bias[512 + f0 + j * 16 + ncol];
    }

    int ntile[6];
#pragma unroll
    for (int j = 0; j < 2; ++j) {
        ntile[j]     = 2 * wave + j;            // z
        ntile[2 + j] = 16 + 2 * wave + j;       // r
        ntile[4 + j] = 32 + 2 * wave + j;       // hhat
    }

    const float* xrow = x + ((size_t)(b0 + ncol) * T_) * D_;

    const int lA = (ncol >= 8) ? 16 : 0;
    const int e0 = ncol & 7;
    const int e1 = 8 + (ncol & 7);

#pragma unroll 1
    for (int t = 0; t < T_; ++t) {
        int tz;                                   // opaque 0: defeats LICM/spilling
        asm volatile("v_mov_b32 %0, 0" : "=v"(tz));

        v8f c[6];
        if constexpr (FUSED) {
#pragma unroll
            for (int j = 0; j < 6; ++j) c[j] = splat8(biasv[j]);
        } else {
            // init accumulators from precomputed xw (bias already folded in)
#pragma unroll
            for (int j = 0; j < 6; ++j) {
                int gcol = ntile[j] * 16 + ncol;
#pragma unroll
                for (int i = 0; i < 8; ++i) {
                    int row = i + 8 * khalf;
                    c[j][i] = xw[((size_t)(b0 + row) * T_ + t) * NG_ + gcol + tz];
                }
            }
        }

        const float* xt = xrow + (size_t)t * D_;
        if constexpr (FUSED) __builtin_prefetch(xt + D_, 0, 1);

        // ---- k-chunks with R resident in LDS ----
#pragma unroll 1
        for (int kc = 0; kc < KLDS; ++kc) {
            v16bf ah = *(const v16bf*)&hbuf[(kc * 32 + lane) * FRAGE + tz];
            v16bf ax;
            if constexpr (FUSED) ax = load_ax(xt, kc, khalf);
#pragma unroll
            for (int j = 0; j < 6; ++j) {
                v16bf brf = *(const v16bf*)&ldsR[((size_t)(ntile[j] * KLDS + kc) * 32 + lane) * FRAGE + tz];
                c[j] = WMMA_BF16(ah, brf, c[j]);
                if constexpr (FUSED) {
                    v16bf bwf = *(const v16bf*)(wsW + ((size_t)(ntile[j] * KC_ + kc) * 32 + lane) * FRAGE + tz);
                    c[j] = WMMA_BF16(ax, bwf, c[j]);
                }
            }
        }
        // ---- k-chunks with R streamed from L2 ----
#pragma unroll 1
        for (int kc = KLDS; kc < KC_; ++kc) {
            v16bf ah = *(const v16bf*)&hbuf[(kc * 32 + lane) * FRAGE + tz];
            v16bf ax;
            if constexpr (FUSED) ax = load_ax(xt, kc, khalf);
#pragma unroll
            for (int j = 0; j < 6; ++j) {
                v16bf brf = *(const v16bf*)(wsR + ((size_t)(ntile[j] * KC_ + kc) * 32 + lane) * FRAGE + tz);
                c[j] = WMMA_BF16(ah, brf, c[j]);
                if constexpr (FUSED) {
                    v16bf bwf = *(const v16bf*)(wsW + ((size_t)(ntile[j] * KC_ + kc) * 32 + lane) * FRAGE + tz);
                    c[j] = WMMA_BF16(ax, bwf, c[j]);
                }
            }
        }

        __syncthreads();   // all waves done reading hbuf

        // ---- gates & state update (wave-local) ----
        v8f z0 = sigmoid8(c[0]), z1 = sigmoid8(c[1]);
        v8f r0 = sigmoid8(c[2]), r1 = sigmoid8(c[3]);
        v8f hh0 = tanh8(r0 * c[4] + (1.0f - r0) * hold0);
        v8f hh1 = tanh8(r1 * c[5] + (1.0f - r1) * hold1);
        v8f hn0 = (1.0f - z0) * hh0 + z0 * hold0;
        v8f hn1 = (1.0f - z1) * hh1 + z1 * hold1;
        hold0 = hn0;
        hold1 = hn1;

        // ---- write h_t (raw; LN pass follows) ----
#pragma unroll
        for (int i = 0; i < 8; ++i) {
            int row = i + 8 * khalf;
            size_t base = ((size_t)(b0 + row) * T_ + t) * U_ + f0 + ncol;
            out[base]      = hn0[i];
            out[base + 16] = hn1[i];
        }

        // ---- scatter h_new (bf16) into k-chunk == wave of hbuf ----
#pragma unroll
        for (int i = 0; i < 8; ++i) {
            int row = i + 8 * khalf;
            hbuf[(wave * 32 + row + lA) * FRAGE + e0] = (__bf16)hn0[i];
            hbuf[(wave * 32 + row + lA) * FRAGE + e1] = (__bf16)hn1[i];
        }
        __syncthreads();
    }
}

// ---------------------------------------------------------------------------
// Kernel 4: in-place LayerNorm, one wave32 per 256-element row.
// ---------------------------------------------------------------------------
__global__ __launch_bounds__(256) void layernorm_inplace(float* __restrict__ y,
                                                         const float* __restrict__ gamma,
                                                         const float* __restrict__ beta) {
    const int wave = threadIdx.x >> 5;
    const int lane = threadIdx.x & 31;
    const size_t row = (size_t)blockIdx.x * 8 + wave;
    float* p = y + row * U_;

    v4f a = *(const v4f*)(p + lane * 8);
    v4f b = *(const v4f*)(p + lane * 8 + 4);

    float s = 0.0f, s2 = 0.0f;
#pragma unroll
    for (int i = 0; i < 4; ++i) {
        s  += a[i] + b[i];
        s2 += a[i] * a[i] + b[i] * b[i];
    }
#pragma unroll
    for (int m = 16; m >= 1; m >>= 1) {
        s  += __shfl_xor(s, m, 32);
        s2 += __shfl_xor(s2, m, 32);
    }
    const float mu  = s * (1.0f / U_);
    const float var = s2 * (1.0f / U_) - mu * mu;
    const float rs  = rsqrtf(var + 1e-6f);

    v4f ga = *(const v4f*)(gamma + lane * 8);
    v4f gb = *(const v4f*)(gamma + lane * 8 + 4);
    v4f ba = *(const v4f*)(beta + lane * 8);
    v4f bb = *(const v4f*)(beta + lane * 8 + 4);
#pragma unroll
    for (int i = 0; i < 4; ++i) {
        a[i] = (a[i] - mu) * rs * ga[i] + ba[i];
        b[i] = (b[i] - mu) * rs * gb[i] + bb[i];
    }
    *(v4f*)(p + lane * 8)     = a;
    *(v4f*)(p + lane * 8 + 4) = b;
}

// ---------------------------------------------------------------------------
extern "C" void kernel_launch(void* const* d_in, const int* in_sizes, int n_in,
                              void* d_out, int out_size, void* d_ws, size_t ws_size,
                              hipStream_t stream) {
    const float* x      = (const float*)d_in[0];   // [B,T,D]
    const float* W      = (const float*)d_in[1];   // [D,3U]
    const float* R      = (const float*)d_in[2];   // [U,3U]
    const float* bias   = (const float*)d_in[3];   // [3U]
    const float* gamma  = (const float*)d_in[4];   // [U]
    const float* beta   = (const float*)d_in[5];   // [U]
    float* out = (float*)d_out;                    // [B,T,U]

    __bf16* wsW = reinterpret_cast<__bf16*>(d_ws);
    __bf16* wsR = wsW + WFRAG;

    const size_t wbytes  = (size_t)2 * WFRAG * sizeof(__bf16);        // 768 KB
    const size_t xwbytes = (size_t)B_ * T_ * NG_ * sizeof(float);     // 384 MiB
    const bool two_phase = ws_size >= wbytes + xwbytes;
    float* xw = (float*)((char*)d_ws + wbytes);

    const size_t smem = 8192 + (size_t)NT_ * KLDS * 32 * FRAGE * sizeof(__bf16); // ~254 KB

    // 1) repack weights to bf16 WMMA fragment layout (L2-resident afterwards)
    prep_weights<<<(WFRAG + 255) / 256, 256, 0, stream>>>(W, R, wsW, wsR);

    if (two_phase) {
        // 2a) whole-chip parallel input projection
        xw_gemm<<<(B_ * T_) / 32, 256, 0, stream>>>(x, bias, wsW, xw);
        // 3a) scan only needs R (mostly LDS-resident)
        gru_scan<false><<<B_ / 16, 256, smem, stream>>>(x, bias, wsW, wsR, xw, out);
    } else {
        // fallback: fused projection + scan
        gru_scan<true><<<B_ / 16, 256, smem, stream>>>(x, bias, wsW, wsR, nullptr, out);
    }

    // 4) in-place LayerNorm, one wave per row
    layernorm_inplace<<<(B_ * T_) / 8, 256, 0, stream>>>(out, gamma, beta);
}
